// AttentionDot_6897717477564
// MI455X (gfx1250) — matrix operand order repaired and checked by reading, not compiled, not run on hardware
//
#include <hip/hip_runtime.h>
#include <hip/hip_bf16.h>

// ---------------------------------------------------------------------------
// AttentionDot forward for MI455X (gfx1250, wave32, WMMA).
//  * softmax over singleton axis == 1.0  -> attn_W/attn_b/prob_W are dead.
//  * conv(attn_ans) is linear in attn_ans = scores * s2 rows:
//      y[b,q,l,c] = sum_k scores[b,q,l+k] * P[b,l+k,k,c],
//      P[b,a,k,c] = sum_d s2[b,a,d]*ctx_Wc[k,d,c]   (batched WMMA GEMM)
//  * sum over q moved before the (linear) C->H projection.
// GEMMs use v_wmma_f32_16x16x32_bf16; each wave computes a 32x32 macro-tile
// (2 A frags x 2 B frags -> 4 WMMA/K-step) to double FLOP/byte vs 16x16.
// Weights are pre-packed into per-(ktile,ntile) fragment order so the B
// fragment load is one contiguous 32B read per lane.
// ---------------------------------------------------------------------------

typedef unsigned short ushortT;
typedef __attribute__((ext_vector_type(16))) __bf16 v16bf;
typedef __attribute__((ext_vector_type(8)))  float  v8f;

static constexpr int Bn = 128, Qn = 32, An = 40, Dn = 300;
static constexpr int Cn = 256, Hn = 512, HLn = 512;

union FragBF { v16bf v; ushortT us[16]; uint2 u2[4]; };

static __device__ __forceinline__ ushortT f2bf(float f) {
  unsigned int u = __float_as_uint(f);
  u += 0x7FFFu + ((u >> 16) & 1u);          // round-to-nearest-even
  return (ushortT)(u >> 16);
}

// ---------------- embedding gather (fp32 -> bf16 rows) ----------------------
__global__ void k_gather(const float* __restrict__ emb, const int* __restrict__ tok,
                         ushortT* __restrict__ dst, long total) {
  long i = (long)blockIdx.x * blockDim.x + threadIdx.x;
  if (i >= total) return;
  long t = i / Dn; int d = (int)(i % Dn);
  dst[i] = f2bf(emb[(long)tok[t] * Dn + d]);
}

// ---------------- scores[b,q,a] = <emb[question], emb[answer]> --------------
__global__ void k_scores(const float* __restrict__ emb, const int* __restrict__ q,
                         const int* __restrict__ a, float* __restrict__ scores) {
  int tid = blockIdx.x * blockDim.x + threadIdx.x;
  if (tid >= Bn * Qn * An) return;
  int b = tid / (Qn * An), r = tid % (Qn * An);
  int qi = r / An, ai = r % An;
  const float* x = emb + (long)q[b * Qn + qi] * Dn;
  const float* y = emb + (long)a[b * An + ai] * Dn;
  float s = 0.f;
  for (int d = 0; d < Dn; ++d) s = fmaf(x[d], y[d], s);
  scores[tid] = s;
}

// ---------------- pack weights fp32[K,N] -> bf16 fragment layout ------------
// dst[((kt*Ntiles+nt)*512) + lane*16 + e]; K = kt*32+(lane<16?0:8)+(e<8?e:e+8), N = nt*16+(lane&15)
__global__ void k_packB(const float* __restrict__ src, int srcK, int srcN,
                        ushortT* __restrict__ dst, int Ntiles, long total) {
  long o = (long)blockIdx.x * blockDim.x + threadIdx.x;
  if (o >= total) return;
  int e = (int)(o & 15), lane = (int)((o >> 4) & 31);
  long t = o >> 9;
  int nt = (int)(t % Ntiles), kt = (int)(t / Ntiles);
  int k = kt * 32 + ((lane & 16) ? 8 : 0) + (e < 8 ? e : e + 8);
  int n = nt * 16 + (lane & 15);
  float v = (k < srcK && n < srcN) ? src[(long)k * srcN + n] : 0.f;
  dst[o] = f2bf(v);
}

// ctx_Wc [K=3,D,C] repacked as B[d, k*C+c] (K-dim = d, padded 300->320), N=768
__global__ void k_packB_ctxWc(const float* __restrict__ Wc, ushortT* __restrict__ dst,
                              long total) {
  long o = (long)blockIdx.x * blockDim.x + threadIdx.x;
  if (o >= total) return;
  int e = (int)(o & 15), lane = (int)((o >> 4) & 31);
  long t = o >> 9;
  int nt = (int)(t % 48), kt = (int)(t / 48);
  int d = kt * 32 + ((lane & 16) ? 8 : 0) + (e < 8 ? e : e + 8);   // K-dim
  int j = nt * 16 + (lane & 15);                  // k*256+c
  float v = (d < Dn) ? Wc[((long)(j >> 8) * Dn + d) * Cn + (j & 255)] : 0.f;
  dst[o] = f2bf(v);
}

// -------- WMMA GEMM, 32x32 macro-tile / wave: C[M,N] = A[M,Kt*32] * Bpack ---
// A: bf16, row m at A + (m/rowsPerSample)*sampleStride + (m%rowsPerSample)*rowStride
// (im2col for the convs is free: window rows are contiguous, stride D).
// Ntiles (16-wide) must be even; M edges handled by row clamp + masked store.
__global__ __launch_bounds__(256)
void k_gemm_wmma(const ushortT* __restrict__ A, int M, int rowsPerSample,
                 long sampleStride, int rowStride, long aBatchStride,
                 const ushortT* __restrict__ Bp, int Ntiles, int Ktiles,
                 float* __restrict__ C, int ldc, int N, long cBatchStride,
                 const float* __restrict__ bias, float biasScale, int Mtiles2) {
  int wave = threadIdx.x >> 5;
  int lane = threadIdx.x & 31;
  long tid = (long)blockIdx.x * 8 + wave;
  int Ntiles2 = Ntiles >> 1;
  long total = (long)Mtiles2 * Ntiles2;
  if (tid >= total) return;                       // wave-uniform
  int mtile2 = (int)(tid % Mtiles2);
  int ntile2 = (int)(tid / Mtiles2);
  A += (long)blockIdx.y * aBatchStride;
  C += (long)blockIdx.y * cBatchStride;

  int mrow0 = mtile2 * 32 + (lane & 15);
  int mrow1 = mrow0 + 16;
  if (mrow0 >= M) mrow0 = M - 1;                  // clamp, store masks later
  if (mrow1 >= M) mrow1 = M - 1;
  const ushortT* rp0 = A + (long)(mrow0 / rowsPerSample) * sampleStride
                         + (long)(mrow0 % rowsPerSample) * rowStride;
  const ushortT* rp1 = A + (long)(mrow1 / rowsPerSample) * sampleStride
                         + (long)(mrow1 % rowsPerSample) * rowStride;
  int kShift = (lane & 16) ? 8 : 0;               // ISA 16-bit A layout
  int ntA = ntile2 * 2, ntB = ntA + 1;

  v8f c00, c01, c10, c11;
  for (int i = 0; i < 8; ++i) { c00[i] = 0.f; c01[i] = 0.f; c10[i] = 0.f; c11[i] = 0.f; }

  for (int kt = 0; kt < Ktiles; ++kt) {
    int kk = kt * 32 + kShift;
    FragBF a0, a1, b0, b1;
    a0.u2[0] = *(const uint2*)(rp0 + kk);         // K = kk .. kk+7
    a0.u2[1] = *(const uint2*)(rp0 + kk + 4);
    a0.u2[2] = *(const uint2*)(rp0 + kk + 16);    // K = kk+16 .. kk+23
    a0.u2[3] = *(const uint2*)(rp0 + kk + 20);
    a1.u2[0] = *(const uint2*)(rp1 + kk);
    a1.u2[1] = *(const uint2*)(rp1 + kk + 4);
    a1.u2[2] = *(const uint2*)(rp1 + kk + 16);
    a1.u2[3] = *(const uint2*)(rp1 + kk + 20);
    const ushortT* bt = Bp + (((long)kt * Ntiles + ntA) << 9) + (lane << 4);
    b0.v = *(const v16bf*)(bt);
    b1.v = *(const v16bf*)(bt + 512);
    c00 = __builtin_amdgcn_wmma_f32_16x16x32_bf16(false, a0.v, false, b0.v, (short)0, c00, false, false);
    c01 = __builtin_amdgcn_wmma_f32_16x16x32_bf16(false, a0.v, false, b1.v, (short)0, c01, false, false);
    c10 = __builtin_amdgcn_wmma_f32_16x16x32_bf16(false, a1.v, false, b0.v, (short)0, c10, false, false);
    c11 = __builtin_amdgcn_wmma_f32_16x16x32_bf16(false, a1.v, false, b1.v, (short)0, c11, false, false);
  }

  int col0  = ntA * 16 + (lane & 15);
  int col1  = col0 + 16;
  int rbase = mtile2 * 32 + ((lane & 16) ? 8 : 0); // ISA f32 C/D layout
  float bv0 = (bias && col0 < N) ? bias[col0] * biasScale : 0.f;
  float bv1 = (bias && col1 < N) ? bias[col1] * biasScale : 0.f;
  for (int v = 0; v < 8; ++v) {
    int r0 = rbase + v, r1 = rbase + 16 + v;
    if (r0 < M && col0 < N) C[(long)r0 * ldc + col0] = c00[v] + bv0;
    if (r0 < M && col1 < N) C[(long)r0 * ldc + col1] = c01[v] + bv1;
    if (r1 < M && col0 < N) C[(long)r1 * ldc + col0] = c10[v] + bv0;
    if (r1 < M && col1 < N) C[(long)r1 * ldc + col1] = c11[v] + bv1;
  }
}

// ---------------- bias + relu + maxpool over time, output bf16 --------------
__global__ void k_pool(const float* __restrict__ y, const float* __restrict__ bc,
                       ushortT* __restrict__ pooled, int L) {
  int tid = blockIdx.x * blockDim.x + threadIdx.x;
  if (tid >= Bn * Cn) return;
  int b = tid / Cn, c = tid % Cn;
  float bcv = bc[c], m = 0.f;                      // relu >= 0 -> init 0 ok
  for (int l = 0; l < L; ++l) {
    float v = y[((long)b * L + l) * Cn + c] + bcv;
    m = fmaxf(m, fmaxf(v, 0.f));
  }
  pooled[tid] = f2bf(m);
}

// ------ y3 = scores * P (3 taps), relu, maxpool over l, SUM over q ----------
__global__ void k_y3pool(const float* __restrict__ scores, const float* __restrict__ P,
                         const float* __restrict__ bc, ushortT* __restrict__ pooled) {
  int tid = blockIdx.x * blockDim.x + threadIdx.x;
  if (tid >= Bn * Cn) return;
  int b = tid / Cn, c = tid % Cn;
  float bcv = bc[c], acc = 0.f;
  const float* Pb = P + (long)b * An * 768 + c;
  for (int q = 0; q < Qn; ++q) {
    const float* sc = scores + ((long)b * Qn + q) * An;
    float m = 0.f;
    for (int l = 0; l < An - 2; ++l) {
      float v = bcv;
      v = fmaf(sc[l + 0], Pb[(long)(l + 0) * 768 + 0],   v);
      v = fmaf(sc[l + 1], Pb[(long)(l + 1) * 768 + 256], v);
      v = fmaf(sc[l + 2], Pb[(long)(l + 2) * 768 + 512], v);
      m = fmaxf(m, fmaxf(v, 0.f));
    }
    acc += m;
  }
  pooled[tid] = f2bf(acc);
}

// ---------------- fp32 -> bf16 convert --------------------------------------
__global__ void k_cvt(const float* __restrict__ src, ushortT* __restrict__ dst, long n) {
  long i = (long)blockIdx.x * blockDim.x + threadIdx.x;
  if (i < n) dst[i] = f2bf(src[i]);
}

// ---------------- batchnorm (batch stats) + tanh ----------------------------
__global__ void k_bn_tanh(const float* __restrict__ x, const float* __restrict__ g,
                          const float* __restrict__ bb, float* __restrict__ out) {
  int h = blockIdx.x * blockDim.x + threadIdx.x;
  if (h >= HLn) return;
  float s = 0.f, s2 = 0.f;
  for (int b = 0; b < Bn; ++b) { float v = x[(long)b * HLn + h]; s += v; s2 += v * v; }
  float mu = s / Bn, var = s2 / Bn - mu * mu;
  float inv = rsqrtf(var + 1e-5f) * g[h], sh = bb[h];
  for (int b = 0; b < Bn; ++b)
    out[(long)b * HLn + h] = tanhf((x[(long)b * HLn + h] - mu) * inv + sh);
}

// ---------------- 512->2 head + log_softmax ---------------------------------
__global__ void k_head(const float* __restrict__ feat, const float* __restrict__ W,
                       const float* __restrict__ bo, float* __restrict__ preds) {
  int b = blockIdx.x * blockDim.x + threadIdx.x;
  if (b >= Bn) return;
  const float* f = feat + (long)b * HLn;
  float l0 = bo[0], l1 = bo[1];
  for (int h = 0; h < HLn; ++h) { float v = f[h]; l0 = fmaf(v, W[h*2], l0); l1 = fmaf(v, W[h*2+1], l1); }
  float m = fmaxf(l0, l1);
  float lse = m + logf(expf(l0 - m) + expf(l1 - m));
  preds[b * 2 + 0] = l0 - lse;
  preds[b * 2 + 1] = l1 - lse;
}

// ---------------------------------------------------------------------------
extern "C" void kernel_launch(void* const* d_in, const int* in_sizes, int n_in,
                              void* d_out, int out_size, void* d_ws, size_t ws_size,
                              hipStream_t stream) {
  (void)in_sizes; (void)n_in; (void)out_size; (void)ws_size;
  const int*   question = (const int*)  d_in[0];
  const int*   answer   = (const int*)  d_in[1];
  const float* emb      = (const float*)d_in[3];
  const float* sm_Wc    = (const float*)d_in[4];   // [3,300,256] == [900,256]
  const float* sm_bc    = (const float*)d_in[5];
  const float* sm_Wf    = (const float*)d_in[6];   // [256,512]
  const float* sm_bf    = (const float*)d_in[7];
  const float* ctx_Wc   = (const float*)d_in[8];
  const float* ctx_bc   = (const float*)d_in[9];
  const float* ctx_Wf   = (const float*)d_in[10];
  const float* ctx_bf   = (const float*)d_in[11];
  const float* fin_W    = (const float*)d_in[15];  // [1536,512]
  const float* fin_b    = (const float*)d_in[16];
  const float* bn_g     = (const float*)d_in[17];
  const float* bn_b     = (const float*)d_in[18];
  const float* out_W    = (const float*)d_in[19];
  const float* out_b    = (const float*)d_in[20];

  // ---- workspace carve-out (256B aligned regions) ----
  char* base = (char*)d_ws;
  size_t off = 0;
  auto carve = [&](size_t bytes) { size_t o = off; off = (off + bytes + 255) & ~(size_t)255; return base + o; };
  ushortT* s1h     = (ushortT*)carve((size_t)(Bn*Qn*Dn + 64) * 2);
  ushortT* s2h     = (ushortT*)carve((size_t)(Bn*An*Dn + 64) * 2);
  ushortT* smWcP   = (ushortT*)carve((size_t)29*16*512 * 2);
  ushortT* ctxWcP  = (ushortT*)carve((size_t)10*48*512 * 2);
  ushortT* smWfP   = (ushortT*)carve((size_t)8*32*512 * 2);
  ushortT* ctxWfP  = (ushortT*)carve((size_t)8*32*512 * 2);
  ushortT* finWP   = (ushortT*)carve((size_t)48*32*512 * 2);
  ushortT* pooled1 = (ushortT*)carve((size_t)Bn*Cn * 2);
  ushortT* pooled2 = (ushortT*)carve((size_t)Bn*Cn * 2);
  ushortT* pooled3 = (ushortT*)carve((size_t)Bn*Cn * 2);
  ushortT* featcatH= (ushortT*)carve((size_t)Bn*3*Hn * 2);
  float*   scores  = (float*)  carve((size_t)Bn*Qn*An * 4);
  float*   y1      = (float*)  carve((size_t)Bn*30*Cn * 4);
  float*   y2      = (float*)  carve((size_t)Bn*38*Cn * 4);
  float*   Pbuf    = (float*)  carve((size_t)Bn*An*768 * 4);
  float*   featcat = (float*)  carve((size_t)Bn*3*Hn * 4);
  float*   featpre = (float*)  carve((size_t)Bn*HLn * 4);

  float* outPreds = (float*)d_out;          // [128,2]
  float* outFeat  = outPreds + Bn * 2;      // [128,512]

  auto cdiv = [](long n, long d) { return (unsigned)((n + d - 1) / d); };

  // ---- gathers + packs + scores ----
  k_gather<<<cdiv((long)Bn*Qn*Dn, 256), 256, 0, stream>>>(emb, question, s1h, (long)Bn*Qn*Dn);
  k_gather<<<cdiv((long)Bn*An*Dn, 256), 256, 0, stream>>>(emb, answer,   s2h, (long)Bn*An*Dn);
  k_packB<<<cdiv((long)29*16*512, 256), 256, 0, stream>>>(sm_Wc, 900, 256, smWcP, 16, (long)29*16*512);
  k_packB_ctxWc<<<cdiv((long)10*48*512, 256), 256, 0, stream>>>(ctx_Wc, ctxWcP, (long)10*48*512);
  k_packB<<<cdiv((long)8*32*512, 256), 256, 0, stream>>>(sm_Wf, 256, 512, smWfP, 32, (long)8*32*512);
  k_packB<<<cdiv((long)8*32*512, 256), 256, 0, stream>>>(ctx_Wf, 256, 512, ctxWfP, 32, (long)8*32*512);
  k_packB<<<cdiv((long)48*32*512, 256), 256, 0, stream>>>(fin_W, 1536, 512, finWP, 32, (long)48*32*512);
  k_scores<<<cdiv((long)Bn*Qn*An, 256), 256, 0, stream>>>(emb, question, answer, scores);

  auto gemm = [&](const ushortT* A, int M, int rps, long sStride, int rStride, long aB,
                  const ushortT* Bp, int Nt, int Kt,
                  float* C, int ldc, int N, long cB,
                  const float* bias, float bScale, int batch) {
    int Mt2 = (M + 31) / 32;
    long total = (long)Mt2 * (Nt >> 1);
    dim3 grid(cdiv(total, 8), batch);
    k_gemm_wmma<<<grid, 256, 0, stream>>>(A, M, rps, sStride, rStride, aB,
                                          Bp, Nt, Kt, C, ldc, N, cB, bias, bScale, Mt2);
  };

  // conv(s1)*sm_Wc : im2col rows contiguous, stride D
  gemm(s1h, Bn*30, 30, (long)Qn*Dn, Dn, 0, smWcP, 16, 29, y1, Cn, Cn, 0, nullptr, 0.f, 1);
  // conv(s2)*sm_Wc
  gemm(s2h, Bn*38, 38, (long)An*Dn, Dn, 0, smWcP, 16, 29, y2, Cn, Cn, 0, nullptr, 0.f, 1);
  // P[b] = s2[b] @ ctx_Wc'  (batched over b)
  gemm(s2h, An, An, 0, Dn, (long)An*Dn, ctxWcP, 48, 10, Pbuf, 768, 768, (long)An*768, nullptr, 0.f, 128);

  // pooling epilogues
  k_pool<<<cdiv(Bn*Cn, 256), 256, 0, stream>>>(y1, sm_bc, pooled1, 30);
  k_pool<<<cdiv(Bn*Cn, 256), 256, 0, stream>>>(y2, sm_bc, pooled2, 38);
  k_y3pool<<<cdiv(Bn*Cn, 256), 256, 0, stream>>>(scores, Pbuf, ctx_bc, pooled3);

  // C->H projections into featcat columns [f1 | f2 | f3sum]; f3 bias scaled by Q
  gemm(pooled1, Bn, Bn, 0, Cn, 0, smWfP,  32, 8, featcat + 0,    3*Hn, Hn, 0, sm_bf, 1.f, 1);
  gemm(pooled2, Bn, Bn, 0, Cn, 0, smWfP,  32, 8, featcat + Hn,   3*Hn, Hn, 0, sm_bf, 1.f, 1);
  gemm(pooled3, Bn, Bn, 0, Cn, 0, ctxWfP, 32, 8, featcat + 2*Hn, 3*Hn, Hn, 0, ctx_bf, (float)Qn, 1);

  k_cvt<<<cdiv((long)Bn*3*Hn, 256), 256, 0, stream>>>(featcat, featcatH, (long)Bn*3*Hn);

  // final linear [128,1536]x[1536,512]
  gemm(featcatH, Bn, Bn, 0, 3*Hn, 0, finWP, 32, 48, featpre, HLn, HLn, 0, fin_b, 1.f, 1);

  k_bn_tanh<<<2, 256, 0, stream>>>(featpre, bn_g, bn_b, outFeat);
  k_head<<<1, 128, 0, stream>>>(outFeat, out_W, out_b, outPreds);
}
